// EnergyPredTransformerGNN_4260607558187
// MI455X (gfx1250) — matrix operand order repaired and checked by reading, not compiled, not run on hardware
//
#include <hip/hip_runtime.h>
#include <hip/hip_bf16.h>
#include <math.h>

#define NN    50000
#define EE    800000
#define GG    64
#define HC    128
#define HEADS 4
#define HD    32

typedef __bf16 bf16_t;
typedef bf16_t v16bf __attribute__((ext_vector_type(16)));
typedef bf16_t v8bf  __attribute__((ext_vector_type(8)));
typedef float  v8f   __attribute__((ext_vector_type(8)));

// ---------------------------------------------------------------------------
// Convert fp32 -> bf16 (4 elements per thread; n must be a multiple of 4).
// ---------------------------------------------------------------------------
__global__ __launch_bounds__(256) void cvt_bf16_kernel(
    const float* __restrict__ in, bf16_t* __restrict__ out, int n4)
{
  int t = blockIdx.x * 256 + threadIdx.x;
  if (t >= n4) return;
  const float4 f = ((const float4*)in)[t];
  bf16_t o[4] = { (bf16_t)f.x, (bf16_t)f.y, (bf16_t)f.z, (bf16_t)f.w };
  ((ulong1*)out)[t] = *(ulong1*)o;   // 8-byte store
}

// Wt[n*Kd + k] = (bf16) W[k*Nc + n]   (transpose + convert; tiny matrices)
__global__ __launch_bounds__(256) void wt_transpose_kernel(
    const float* __restrict__ W, bf16_t* __restrict__ Wt, int Kd, int Nc)
{
  int t = blockIdx.x * 256 + threadIdx.x;
  if (t >= Kd * Nc) return;
  const int n = t / Kd, k = t - n * Kd;
  Wt[t] = (bf16_t)W[(size_t)k * Nc + n];
}

// ---------------------------------------------------------------------------
// Dense linear:  Out[M,Nc] = act( X[M,Kd] @ W[Kd,Nc] + bias[Nc] )
// Xb: bf16 row-major [M,Kd].  Wt: bf16 TRANSPOSED [Nc,Kd].
// v_wmma_f32_16x16x32_bf16, f32 accumulate.
// Block = 128 threads = 4 waves; wave w owns a 16x32 output tile
// (one A fragment, two B fragments / accumulators), K unrolled x4.
// Requires M % 16 == 0 (call sites: 50000, 64).
// Fragment packing (this code's contiguous k order, same in A and B):
//   A: lane row m = l16, elem e -> K = half*16+e
//      => two contiguous 8-bf16 (16 B) runs.
//   B: lane col n, elem e -> K = half*16+e => 16 contiguous bf16 (32 B) in Wt.
//   C/D: VGPR v -> M = v + 8*half, N = l16.
// ---------------------------------------------------------------------------
__global__ __launch_bounds__(128) void wmma_linear_kernel(
    const bf16_t* __restrict__ Xb, const bf16_t* __restrict__ Wt,
    const float* __restrict__ bias, float* __restrict__ Out,
    int M, int Kd, int Nc, int relu)
{
  const int lane = threadIdx.x & 31;
  const int wave = threadIdx.x >> 5;
  const int half = lane >> 4;
  const int l16  = lane & 15;
  const int colbase = wave * 32;
  if (colbase >= Nc) return;       // wave-uniform skip (EXEC stays all-ones)
  const int rowbase = blockIdx.x * 16;

  const int n0  = colbase + l16;
  const int n1  = colbase + 16 + l16;
  const int nn0 = (n0 < Nc) ? n0 : (Nc - 1);       // clamped for loads
  const int nn1 = (n1 < Nc) ? n1 : (Nc - 1);
  const int m   = rowbase + l16;

  const bf16_t* xrow = Xb + (size_t)m * Kd + half * 16;
  const bf16_t* wrow0 = Wt + (size_t)nn0 * Kd + half * 16;
  const bf16_t* wrow1 = Wt + (size_t)nn1 * Kd + half * 16;

  v8f acc0 = {};
  v8f acc1 = {};
#pragma unroll 4
  for (int kk = 0; kk < Kd; kk += 32) {
    const v8bf alo = *(const v8bf*)(xrow + kk);        // K = kk+half*16 .. +7
    const v8bf ahi = *(const v8bf*)(xrow + kk + 8);    // K = kk+half*16+8 ..
    v16bf A;
#pragma unroll
    for (int e = 0; e < 8; e++) { A[e] = alo[e]; A[8 + e] = ahi[e]; }

    const v16bf B0 = *(const v16bf*)(wrow0 + kk);      // K = kk+half*16 .. +15
    const v16bf B1 = *(const v16bf*)(wrow1 + kk);

    acc0 = __builtin_amdgcn_wmma_f32_16x16x32_bf16(
        false, A, false, B0, (short)0, acc0, false, false);
    acc1 = __builtin_amdgcn_wmma_f32_16x16x32_bf16(
        false, A, false, B1, (short)0, acc1, false, false);
  }

  if (n0 < Nc) {
    const float bv = bias[n0];
    float* op = Out + (size_t)(rowbase + 8 * half) * Nc + n0;
#pragma unroll
    for (int vv = 0; vv < 8; vv++) {
      float o = acc0[vv] + bv;
      if (relu) o = fmaxf(o, 0.0f);
      op[(size_t)vv * Nc] = o;
    }
  }
  if (n1 < Nc) {
    const float bv = bias[n1];
    float* op = Out + (size_t)(rowbase + 8 * half) * Nc + n1;
#pragma unroll
    for (int vv = 0; vv < 8; vv++) {
      float o = acc1[vv] + bv;
      if (relu) o = fmaxf(o, 0.0f);
      op[(size_t)vv * Nc] = o;
    }
  }
}

// ---------------------------------------------------------------------------
__global__ __launch_bounds__(256) void fill_kernel(float* p, float val, int n)
{
  int t = blockIdx.x * 256 + threadIdx.x;
  if (t < n) p[t] = val;
}

__device__ __forceinline__ void atomicMaxF(float* addr, float val)
{
  int* ai = (int*)addr;
  int cur = __float_as_int(*addr);
  while (val > __int_as_float(cur)) {
    int prev = atomicCAS(ai, cur, __float_as_int(val));
    if (prev == cur) break;
    cur = prev;
  }
}

// alpha[e,h] = <q[dst], k[src] + ew*We> / sqrt(HD);  segment max into amax.
__global__ __launch_bounds__(256) void edge_alpha_kernel(
    const float* __restrict__ q, const float* __restrict__ k,
    const float* __restrict__ We, const int* __restrict__ src,
    const int* __restrict__ dst, const float* __restrict__ ew,
    float* __restrict__ alpha, float* __restrict__ amax)
{
  int t = blockIdx.x * 256 + threadIdx.x;
  if (t >= EE * HEADS) return;
  const int e = t >> 2, h = t & 3;
  const int s = src[e], d = dst[e];
  const float w = ew[e];
  const float4* qp = (const float4*)(q + (size_t)d * HC + h * HD);
  const float4* kp = (const float4*)(k + (size_t)s * HC + h * HD);
  const float4* wp = (const float4*)(We + h * HD);
  float acc = 0.0f;
#pragma unroll
  for (int i = 0; i < 8; i++) {
    const float4 qv = qp[i], kv = kp[i], wv = wp[i];
    acc += qv.x * (kv.x + w * wv.x) + qv.y * (kv.y + w * wv.y)
         + qv.z * (kv.z + w * wv.z) + qv.w * (kv.w + w * wv.w);
  }
  const float a = acc * 0.17677669529663687f;  // 1/sqrt(32)
  alpha[t] = a;
  atomicMaxF(&amax[d * HEADS + h], a);
}

// ex = exp(alpha - amax[dst]); alpha <- ex; denom[dst] += ex.
__global__ __launch_bounds__(256) void edge_exp_kernel(
    float* __restrict__ alpha, const float* __restrict__ amax,
    const int* __restrict__ dst, float* __restrict__ denom)
{
  int t = blockIdx.x * 256 + threadIdx.x;
  if (t >= EE * HEADS) return;
  const int e = t >> 2, h = t & 3;
  const int d = dst[e];
  const float ex = __expf(alpha[t] - amax[d * HEADS + h]);
  alpha[t] = ex;
  atomicAdd(&denom[d * HEADS + h], ex);
}

// aggr[dst,c] += (v[src,c] + ew*We[c]) * ex / (denom[dst]+1e-16)
__global__ __launch_bounds__(256) void edge_scatter_kernel(
    const float* __restrict__ v, const float* __restrict__ We,
    const float* __restrict__ alpha, const float* __restrict__ denom,
    const int* __restrict__ src, const int* __restrict__ dst,
    const float* __restrict__ ew, float* __restrict__ aggr)
{
  int t = blockIdx.x * 256 + threadIdx.x;
  if (t >= EE * HC) return;
  const int e = t >> 7, c = t & 127, h = c >> 5;
  const int s = src[e], d = dst[e];
  const float coef = alpha[e * HEADS + h] / (denom[d * HEADS + h] + 1e-16f);
  const float val = (v[(size_t)s * HC + c] + ew[e] * We[c]) * coef;
  atomicAdd(&aggr[(size_t)d * HC + c], val);
}

// y = aggr + skip (in place into aggr); per-channel sum / sumsq via LDS reduce.
__global__ __launch_bounds__(256) void skip_stats_kernel(
    float* __restrict__ yio, const float* __restrict__ sk,
    float* __restrict__ stats, int total)
{
  __shared__ float ss[256];
  __shared__ float sq[256];
  float ps = 0.0f, pq = 0.0f;
  const int stride = gridDim.x * 256;  // multiple of 128: column stays fixed
  for (int t = blockIdx.x * 256 + threadIdx.x; t < total; t += stride) {
    const float yv = yio[t] + sk[t];
    yio[t] = yv;
    ps += yv; pq += yv * yv;
  }
  ss[threadIdx.x] = ps; sq[threadIdx.x] = pq;
  __syncthreads();
  if (threadIdx.x < 128) {
    atomicAdd(&stats[threadIdx.x],       ss[threadIdx.x] + ss[threadIdx.x + 128]);
    atomicAdd(&stats[128 + threadIdx.x], sq[threadIdx.x] + sq[threadIdx.x + 128]);
  }
}

__global__ void bn_finalize_kernel(const float* __restrict__ stats,
                                   const float* __restrict__ g,
                                   const float* __restrict__ b,
                                   float* __restrict__ ssh)
{
  const int c = threadIdx.x;
  if (c < HC) {
    const float m   = stats[c] * (1.0f / NN);
    const float var = stats[HC + c] * (1.0f / NN) - m * m;
    const float sc  = g[c] * rsqrtf(var + 1e-5f);
    ssh[c]      = sc;
    ssh[HC + c] = b[c] - m * sc;
  }
}

__global__ __launch_bounds__(256) void bn_relu_kernel(
    const float* __restrict__ y, const float* __restrict__ ssh,
    float* __restrict__ out, int total)
{
  int t = blockIdx.x * 256 + threadIdx.x;
  if (t >= total) return;
  const int c = t & 127;
  out[t] = fmaxf(y[t] * ssh[c] + ssh[HC + c], 0.0f);
}

__global__ __launch_bounds__(256) void pool_kernel(
    const float* __restrict__ h, const int* __restrict__ batch,
    float* __restrict__ gsum)
{
  int t = blockIdx.x * 256 + threadIdx.x;
  if (t >= NN * HC) return;
  const int i = t >> 7, c = t & 127;
  atomicAdd(&gsum[batch[i] * HC + c], h[t]);
}

__global__ __launch_bounds__(256) void cnt_kernel(
    const int* __restrict__ batch, float* __restrict__ gcnt)
{
  int t = blockIdx.x * 256 + threadIdx.x;
  if (t >= NN) return;
  atomicAdd(&gcnt[batch[t]], 1.0f);
}

// comb[g, 0:128] = gsum/max(cnt,1);  comb[g, 128:256] = ie[g]*Wi + bi
__global__ __launch_bounds__(256) void comb_kernel(
    const float* __restrict__ gsum, const float* __restrict__ gcnt,
    const float* __restrict__ ie, const float* __restrict__ Wi,
    const float* __restrict__ bi, float* __restrict__ comb)
{
  int t = blockIdx.x * 256 + threadIdx.x;
  if (t >= GG * 2 * HC) return;
  const int g = t >> 8, c = t & 255;
  float o;
  if (c < HC) o = gsum[g * HC + c] / fmaxf(gcnt[g], 1.0f);
  else { const int cc = c - HC; o = ie[g] * Wi[cc] + bi[cc]; }
  comb[t] = o;
}

// ---------------------------------------------------------------------------
extern "C" void kernel_launch(void* const* d_in, const int* in_sizes, int n_in,
                              void* d_out, int out_size, void* d_ws, size_t ws_size,
                              hipStream_t stream)
{
  const float* x     = (const float*)d_in[0];
  const int*   ei    = (const int*)d_in[1];
  const int*   src   = ei;
  const int*   dst   = ei + EE;
  const float* ew    = (const float*)d_in[2];
  const int*   batch = (const int*)d_in[3];
  const float* ie    = (const float*)d_in[4];

  // params flattened in dict-insertion order starting at index 5
  // layer i base = 5 + 11*i : Wq bq Wk bk Wv bv We Ws bs | bn: g b
  const float* Wi  = (const float*)d_in[38];
  const float* bi  = (const float*)d_in[39];
  const float* W1  = (const float*)d_in[40];
  const float* b1  = (const float*)d_in[41];
  const float* W2  = (const float*)d_in[42];
  const float* b2  = (const float*)d_in[43];

  // workspace layout (float-granularity allocator; keep 16B alignment)
  size_t off = 0;
  auto alloc = [&](size_t nfl) {
    float* p = (float*)((char*)d_ws + off);
    off += ((nfl * sizeof(float) + 15) & ~(size_t)15);
    return p;
  };
  float*  hbuf  = alloc((size_t)NN * HC);
  float*  qb    = alloc((size_t)NN * HC);
  float*  kb    = alloc((size_t)NN * HC);
  float*  vb    = alloc((size_t)NN * HC);
  float*  skb   = alloc((size_t)NN * HC);
  float*  aggr  = alloc((size_t)NN * HC);   // also holds y = aggr + skip
  float*  alpha = alloc((size_t)EE * HEADS);
  float*  amax  = alloc((size_t)NN * HEADS);
  float*  denom = alloc((size_t)NN * HEADS);
  float*  stats = alloc(2 * HC);
  float*  ssh   = alloc(2 * HC);
  float*  gsum  = alloc((size_t)GG * HC);
  float*  gcnt  = alloc(GG);
  float*  comb  = alloc((size_t)GG * 2 * HC);
  float*  hcomb = alloc((size_t)GG * HC);
  bf16_t* xb    = (bf16_t*)alloc((size_t)NN * HC / 2);      // bf16 activations
  bf16_t* wt    = (bf16_t*)alloc((size_t)2 * HC * HC / 2);  // bf16 W^T scratch
  bf16_t* combb = (bf16_t*)alloc((size_t)GG * 2 * HC / 2);
  (void)ws_size; (void)in_sizes; (void)n_in; (void)out_size;

  const int gemmBlocks = NN / 16;                 // 50000 = 3125 * 16 exactly
  const int eh   = (EE * HEADS + 255) / 256;
  const int ehc  = (EE * HC + 255) / 256;
  const int nhc  = (NN * HC + 255) / 256;
  const int nhc4 = (NN * HC / 4 + 255) / 256;

  const float* hin = x;
  for (int i = 0; i < 3; i++) {
    const int base = 5 + 11 * i;
    const float* Wq = (const float*)d_in[base + 0];
    const float* bq = (const float*)d_in[base + 1];
    const float* Wk = (const float*)d_in[base + 2];
    const float* bk = (const float*)d_in[base + 3];
    const float* Wv = (const float*)d_in[base + 4];
    const float* bv = (const float*)d_in[base + 5];
    const float* We = (const float*)d_in[base + 6];
    const float* Ws = (const float*)d_in[base + 7];
    const float* bs = (const float*)d_in[base + 8];
    const float* bng = (const float*)d_in[base + 9];
    const float* bnb = (const float*)d_in[base + 10];

    // activations -> bf16 once per layer
    cvt_bf16_kernel<<<nhc4, 256, 0, stream>>>(hin, xb, NN * HC / 4);

    // 4 projections; weight transpose+convert reuses one scratch slot
    // (stream order serializes transpose -> gemm).
    const float* Wmat[4] = { Wq, Wk, Wv, Ws };
    const float* bvec[4] = { bq, bk, bv, bs };
    float*       outp[4] = { qb, kb, vb, skb };
    for (int j = 0; j < 4; j++) {
      wt_transpose_kernel<<<(HC * HC + 255) / 256, 256, 0, stream>>>(Wmat[j], wt, HC, HC);
      wmma_linear_kernel<<<gemmBlocks, 128, 0, stream>>>(xb, wt, bvec[j], outp[j],
                                                         NN, HC, HC, 0);
    }

    fill_kernel<<<(NN * HEADS + 255) / 256, 256, 0, stream>>>(amax, -__builtin_inff(), NN * HEADS);
    fill_kernel<<<(NN * HEADS + 255) / 256, 256, 0, stream>>>(denom, 0.0f, NN * HEADS);
    fill_kernel<<<nhc, 256, 0, stream>>>(aggr, 0.0f, NN * HC);
    fill_kernel<<<1, 256, 0, stream>>>(stats, 0.0f, 2 * HC);

    edge_alpha_kernel<<<eh, 256, 0, stream>>>(qb, kb, We, src, dst, ew, alpha, amax);
    edge_exp_kernel<<<eh, 256, 0, stream>>>(alpha, amax, dst, denom);
    edge_scatter_kernel<<<ehc, 256, 0, stream>>>(vb, We, alpha, denom, src, dst, ew, aggr);

    skip_stats_kernel<<<512, 256, 0, stream>>>(aggr, skb, stats, NN * HC);
    bn_finalize_kernel<<<1, 128, 0, stream>>>(stats, bng, bnb, ssh);
    bn_relu_kernel<<<nhc, 256, 0, stream>>>(aggr, ssh, hbuf, NN * HC);
    hin = hbuf;
  }

  fill_kernel<<<(GG * HC + 255) / 256, 256, 0, stream>>>(gsum, 0.0f, GG * HC);
  fill_kernel<<<1, 256, 0, stream>>>(gcnt, 0.0f, GG);
  pool_kernel<<<nhc, 256, 0, stream>>>(hin, batch, gsum);
  cnt_kernel<<<(NN + 255) / 256, 256, 0, stream>>>(batch, gcnt);
  comb_kernel<<<(GG * 2 * HC + 255) / 256, 256, 0, stream>>>(gsum, gcnt, ie, Wi, bi, comb);

  // fc1: [64,256]@[256,128]+b1, ReLU
  cvt_bf16_kernel<<<(GG * 2 * HC / 4 + 255) / 256, 256, 0, stream>>>(comb, combb, GG * 2 * HC / 4);
  wt_transpose_kernel<<<(2 * HC * HC + 255) / 256, 256, 0, stream>>>(W1, wt, 2 * HC, HC);
  wmma_linear_kernel<<<GG / 16, 128, 0, stream>>>(combb, wt, b1, hcomb, GG, 2 * HC, HC, 1);

  // fc2: [64,128]@[128,100]+b2  (reuse combb as bf16 staging for hcomb)
  cvt_bf16_kernel<<<(GG * HC / 4 + 255) / 256, 256, 0, stream>>>(hcomb, combb, GG * HC / 4);
  wt_transpose_kernel<<<(HC * 100 + 255) / 256, 256, 0, stream>>>(W2, wt, HC, 100);
  wmma_linear_kernel<<<GG / 16, 128, 0, stream>>>(combb, wt, b2, (float*)d_out, GG, HC, 100, 0);
}
